// ContentPredictor_71519795413211
// MI455X (gfx1250) — matrix operand order repaired and checked by reading, not compiled, not run on hardware
//
#include <hip/hip_runtime.h>
#include <hip/hip_bf16.h>

typedef __attribute__((ext_vector_type(2)))  float  v2f;
typedef __attribute__((ext_vector_type(8)))  float  v8f;
typedef __attribute__((ext_vector_type(4)))  int    v4i;
typedef __attribute__((ext_vector_type(8)))  __bf16 v8bf;
typedef __attribute__((ext_vector_type(16))) __bf16 v16bf;

#define ZC 512
#define D  256
#define K  8192
#define NROW 32768            // B*T = 8*4096
#define NELEM (NROW * D)      // 8388608
#define CT 64                 // codes staged per LDS chunk (32 KB)
#define NCHUNK (K / CT)       // 128

#if __has_builtin(__builtin_amdgcn_global_load_async_to_lds_b128)
#define ASYNC_PATH 1
#else
#define ASYNC_PATH 0
#endif

__device__ __forceinline__ unsigned short f32_to_bf16_rne(float v) {
  unsigned int u = __builtin_bit_cast(unsigned int, v);
  unsigned int r = (u + 0x7FFFu + ((u >> 16) & 1u)) >> 16;
  return (unsigned short)r;
}

__device__ __forceinline__ void wait_async0() {
#if ASYNC_PATH
#if __has_builtin(__builtin_amdgcn_s_wait_asynccnt)
  __builtin_amdgcn_s_wait_asynccnt(0);
#else
  asm volatile("s_wait_asynccnt 0x0" ::: "memory");
#endif
#endif
}

// Stage CT codebook rows (CT*D bf16 = 32 KB) into LDS. 256 threads, 8 x b128 each.
__device__ __forceinline__ void issue_chunk(const unsigned short* __restrict__ cbf,
                                            unsigned short* buf, int n0, int tid) {
#if ASYNC_PATH
  v4i __attribute__((address_space(1)))* g =
      (v4i __attribute__((address_space(1)))*)(cbf + (size_t)n0 * D);
  v4i __attribute__((address_space(3)))* l =
      (v4i __attribute__((address_space(3)))*)buf;
#pragma unroll
  for (int r = 0; r < 8; ++r) {
    const int e = r * 256 + tid;            // 16B units; 4 KB/round, 8 rounds
    __builtin_amdgcn_global_load_async_to_lds_b128(g + e, l + e, 0, 0);
  }
#else
  const v4i* g = (const v4i*)(cbf + (size_t)n0 * D);
  v4i* l = (v4i*)buf;
#pragma unroll
  for (int r = 0; r < 8; ++r) l[r * 256 + tid] = g[r * 256 + tid];
#endif
}

// ---------------------------------------------------------------------------
// K1: h = x @ W + b   (f32 WMMA 16x16x4). One wave per 16x16 output tile.
// ---------------------------------------------------------------------------
__global__ void __launch_bounds__(128) hgemm_kernel(
    const float* __restrict__ x, const float* __restrict__ W,
    const float* __restrict__ bias, float* __restrict__ h,
    unsigned short* __restrict__ hbf) {
  const int lane = threadIdx.x & 31;
  const int wave = threadIdx.x >> 5;
  const int wg   = blockIdx.x * 4 + wave;
  const int tileM = (wg >> 4) * 16;     // 2048 M-tiles
  const int tileN = (wg & 15) * 16;     // 16 N-tiles
  const int half = lane >> 4;
  const int l16  = lane & 15;

  const float* xrow = x + (size_t)(tileM + l16) * ZC;
  v8f acc = {};
#pragma unroll 4
  for (int k = 0; k < ZC / 4; ++k) {
    const int kb = k * 4 + 2 * half;    // A: lane holds K = kb, kb+1 (contig b64)
    v2f a; a.x = xrow[kb]; a.y = xrow[kb + 1];
    v2f b;                              // B: col = l16, rows kb, kb+1
    b.x = W[(size_t)kb * D + tileN + l16];
    b.y = W[(size_t)(kb + 1) * D + tileN + l16];
    acc = __builtin_amdgcn_wmma_f32_16x16x4_f32(
        false, a, false, b, (short)0, acc, false, false);
  }
  const float bv = bias[tileN + l16];
#pragma unroll
  for (int j = 0; j < 8; ++j) {
    const int row = tileM + j + half * 8;
    const int col = tileN + l16;
    const float v = acc[j] + bv;
    h[(size_t)row * D + col]   = v;
    hbf[(size_t)row * D + col] = f32_to_bf16_rne(v);
  }
}

// ---------------------------------------------------------------------------
// K2: cnorm[k] = ||codebook[k]||^2 (f32) + bf16 copy. One wave per code row.
// ---------------------------------------------------------------------------
__global__ void __launch_bounds__(256) cnorm_kernel(
    const float* __restrict__ cb, float* __restrict__ cnorm,
    unsigned short* __restrict__ cbf) {
  const int lane = threadIdx.x & 31;
  const int wave = threadIdx.x >> 5;
  const int row  = blockIdx.x * 8 + wave;      // 1024 blocks -> 8192 rows
  const float* r = cb + (size_t)row * D;
  float s = 0.0f;
#pragma unroll
  for (int i = lane; i < D; i += 32) {
    const float v = r[i];
    s += v * v;
    cbf[(size_t)row * D + i] = f32_to_bf16_rne(v);
  }
#pragma unroll
  for (int off = 16; off; off >>= 1) s += __shfl_xor(s, off, 32);
  if (lane == 0) cnorm[row] = s;
}

// ---------------------------------------------------------------------------
// K3: argmin_k ( ||e_k||^2 - 2 h.e_k )  via bf16 WMMA from LDS-staged codebook.
// 8 waves/block, each wave owns a 16-row tile; codebook double-buffered in LDS
// via async load-to-LDS (ASYNCcnt), 64 codes per chunk.
// ---------------------------------------------------------------------------
__global__ void __launch_bounds__(256, 2) argmin_kernel(
    const unsigned short* __restrict__ hbf,
    const unsigned short* __restrict__ cbf,
    const float* __restrict__ cnorm,
    int* __restrict__ idxOut) {
  __shared__ unsigned short sbuf[2][CT * D];   // 2 x 32 KB

  const int tid   = threadIdx.x;
  const int lane  = tid & 31;
  const int wave  = tid >> 5;
  const int tileM = (blockIdx.x * 8 + wave) * 16;   // 256 blocks * 8 waves
  const int half  = lane >> 4;
  const int l16   = lane & 15;

  // Prologue: start streaming chunk 0 while we build A fragments.
  issue_chunk(cbf, sbuf[0], 0, tid);

  // A-fragments: row = tileM + l16; 8 fragments cover D=256 (K=32 each).
  v16bf a[8];
  {
    const unsigned short* hr = hbf + (size_t)(tileM + l16) * D;
#pragma unroll
    for (int f = 0; f < 8; ++f) {
      const int d0 = f * 32 + half * 8;        // elems 0..7 : K 0..7 (+8)
      const int d1 = f * 32 + 16 + half * 8;   // elems 8..15: K 16..23 (+8)
      const v8bf lo = *(const v8bf*)(hr + d0);
      const v8bf hi = *(const v8bf*)(hr + d1);
#pragma unroll
      for (int t = 0; t < 8; ++t) { a[f][t] = lo[t]; a[f][t + 8] = hi[t]; }
    }
  }

  float minv[8]; int mini[8];
#pragma unroll
  for (int j = 0; j < 8; ++j) { minv[j] = 3.4028235e38f; mini[j] = 0x7FFFFFFF; }

  for (int c = 0; c < NCHUNK; ++c) {
    wait_async0();          // chunk c resident (this wave's portion)
    __syncthreads();        // all waves' portions resident; other buffer free
    if (c + 1 < NCHUNK) issue_chunk(cbf, sbuf[(c + 1) & 1], (c + 1) * CT, tid);

    const unsigned short* lb = sbuf[c & 1];
#pragma unroll
    for (int t = 0; t < CT / 16; ++t) {        // 4 code tiles per chunk
      const int n = c * CT + t * 16;
      const unsigned short* crow = lb + (size_t)(t * 16 + l16) * D;
      const float cn = cnorm[n + l16];
      v8f acc = {};
#pragma unroll
      for (int f = 0; f < 8; ++f) {
        // B: elems 0..7 -> K 8h..8h+7, elems 8..15 -> K 16+8h..16+8h+7
        const v8bf blo = *(const v8bf*)(crow + f * 32 + half * 8);
        const v8bf bhi = *(const v8bf*)(crow + f * 32 + 16 + half * 8);
        const v16bf bfrag = __builtin_shufflevector(blo, bhi, 0, 1, 2, 3, 4, 5, 6, 7, 8, 9, 10, 11, 12, 13, 14, 15);
        acc = __builtin_amdgcn_wmma_f32_16x16x32_bf16(
            false, a[f], false, bfrag, (short)0, acc, false, false);
      }
#pragma unroll
      for (int j = 0; j < 8; ++j) {
        const float s = __builtin_fmaf(-2.0f, acc[j], cn);
        if (s < minv[j]) { minv[j] = s; mini[j] = n + l16; }
      }
    }
  }

  // Cross-lane min over the 16 lanes of each half (xor <16 stays in half).
#pragma unroll
  for (int off = 1; off < 16; off <<= 1) {
#pragma unroll
    for (int j = 0; j < 8; ++j) {
      const float ov = __shfl_xor(minv[j], off, 32);
      const int   oi = __shfl_xor(mini[j], off, 32);
      if (ov < minv[j] || (ov == minv[j] && oi < mini[j])) {
        minv[j] = ov; mini[j] = oi;
      }
    }
  }
  if (l16 == 0) {
#pragma unroll
    for (int j = 0; j < 8; ++j) idxOut[tileM + j + half * 8] = mini[j];
  }
}

// ---------------------------------------------------------------------------
// K4: quant = codebook[idx]; per-row sum of (zq - h)^2 -> partials[row].
// ---------------------------------------------------------------------------
__global__ void __launch_bounds__(256) gather_loss_kernel(
    const float* __restrict__ h, const float* __restrict__ cb,
    const int* __restrict__ idx, float* __restrict__ quant,
    float* __restrict__ partials) {
  __shared__ float red[256];
  const int row = blockIdx.x;
  const int d   = threadIdx.x;
  const int code = idx[row];
  const float zq = cb[(size_t)code * D + d];
  const float hv = h[(size_t)row * D + d];
  quant[(size_t)row * D + d] = zq;
  const float diff = zq - hv;
  red[d] = diff * diff;
  __syncthreads();
#pragma unroll
  for (int s = 128; s; s >>= 1) {
    if (d < s) red[d] += red[d + s];
    __syncthreads();
  }
  if (d == 0) partials[row] = red[0];
}

// ---------------------------------------------------------------------------
// K5: deterministic reduction of 32768 partials -> loss scalar.
// ---------------------------------------------------------------------------
__global__ void __launch_bounds__(256) finalize_kernel(
    const float* __restrict__ partials, float* __restrict__ lossOut) {
  __shared__ float red[256];
  float s = 0.0f;
  for (int i = threadIdx.x; i < NROW; i += 256) s += partials[i];
  red[threadIdx.x] = s;
  __syncthreads();
#pragma unroll
  for (int t = 128; t; t >>= 1) {
    if (threadIdx.x < t) red[threadIdx.x] += red[threadIdx.x + t];
    __syncthreads();
  }
  if (threadIdx.x == 0)
    lossOut[0] = 1.25f * red[0] / (float)NELEM;   // (1+BETA)*mean
}

// ---------------------------------------------------------------------------
extern "C" void kernel_launch(void* const* d_in, const int* in_sizes, int n_in,
                              void* d_out, int out_size, void* d_ws, size_t ws_size,
                              hipStream_t stream) {
  const float* x  = (const float*)d_in[0];   // [8,4096,1,512]
  const float* W  = (const float*)d_in[1];   // [512,256]
  const float* b  = (const float*)d_in[2];   // [256]
  const float* cb = (const float*)d_in[3];   // [8192,256]
  float* out = (float*)d_out;                // [NELEM] quant + [1] loss

  char* ws = (char*)d_ws;
  float*          h_f32    = (float*)(ws);                         // 32 MB
  unsigned short* h_bf16   = (unsigned short*)(ws + 33554432);     // 16 MB
  unsigned short* cb_bf16  = (unsigned short*)(ws + 50331648);     //  4 MB
  float*          cnorm    = (float*)(ws + 54525952);              // 32 KB
  int*            idx      = (int*)(ws + 54558720);                // 128 KB
  float*          partials = (float*)(ws + 54689792);              // 128 KB

  hipLaunchKernelGGL(hgemm_kernel, dim3(8192), dim3(128), 0, stream,
                     x, W, b, h_f32, h_bf16);
  hipLaunchKernelGGL(cnorm_kernel, dim3(1024), dim3(256), 0, stream,
                     cb, cnorm, cb_bf16);
  hipLaunchKernelGGL(argmin_kernel, dim3(256), dim3(256), 0, stream,
                     h_bf16, cb_bf16, cnorm, idx);
  hipLaunchKernelGGL(gather_loss_kernel, dim3(NROW), dim3(256), 0, stream,
                     h_f32, cb, idx, out, partials);
  hipLaunchKernelGGL(finalize_kernel, dim3(1), dim3(256), 0, stream,
                     partials, out + NELEM);
}
